// UV_Aggregator_19061064860210
// MI455X (gfx1250) — matrix operand order repaired and checked by reading, not compiled, not run on hardware
//
#include <hip/hip_runtime.h>

typedef __attribute__((ext_vector_type(16))) _Float16 v16h;
typedef __attribute__((ext_vector_type(8)))  float    v8f;

#define EMBED 64
#define HL 50

union V16H { v16h v; _Float16 h[16]; uint4 q[2]; };
union V8F  { v8f  v; float    f[8];  };

// -----------------------------------------------------------------------------
// Pre-kernel: convert an [Kdim x 64] f32 weight (row-major, in-major) into the
// exact per-lane WMMA B-operand order for v_wmma_f32_16x16x32_f16.
// B tile (kb,nt): lane L holds column n = nt*16 + (L&15); lane group g = L>>4
// holds K = kb*32 + g*8 + {0..7} and kb*32 + 16 + g*8 + {0..7}; VGPR v packs elements 2v, 2v+1 (low/high half).
// Flat pair index t = kb*1024 + nt*256 + lane*8 + v  ->  out[2t], out[2t+1].
// -----------------------------------------------------------------------------
__global__ void cvt_weights(const float* __restrict__ W,
                            _Float16* __restrict__ out, int npairs) {
    int t = blockIdx.x * blockDim.x + threadIdx.x;
    if (t >= npairs) return;
    int v    = t & 7;
    int lane = (t >> 3) & 31;
    int nt   = (t >> 8) & 3;
    int kb   = t >> 10;
    int n    = nt * 16 + (lane & 15);
    int g    = lane >> 4;
    int k0   = kb * 32 + g * 8 + 2 * v + ((v >= 4) ? 8 : 0);
    out[2 * t]     = (_Float16)W[k0 * EMBED + n];
    out[2 * t + 1] = (_Float16)W[(k0 + 1) * EMBED + n];
}

// -----------------------------------------------------------------------------
// Main kernel: one block per batch element; 4 waves, each owning a 16-row tile
// of the 64-padded history dimension.
// -----------------------------------------------------------------------------
__global__ __launch_bounds__(128)
void uv_agg(const int* __restrict__ nodes, const int* __restrict__ hist_uv,
            const int* __restrict__ hist_r, const int* __restrict__ lengths,
            const float* __restrict__ u2e, const float* __restrict__ v2e,
            const float* __restrict__ r2e,
            const _Float16* __restrict__ w1h, const float* __restrict__ b1,
            const _Float16* __restrict__ w2h, const float* __restrict__ b2,
            const _Float16* __restrict__ a1h, const float* __restrict__ ba1,
            const _Float16* __restrict__ a2h, const float* __restrict__ ba2,
            const float* __restrict__ a3, const float* __restrict__ ba3,
            float* __restrict__ out)
{
    __shared__ _Float16 sh_O[4 * 16 * EMBED];      // o_history (f16), persists
    __shared__ _Float16 sh_stage[4 * 16 * EMBED];  // inter-layer staging
    __shared__ _Float16 sh_uvh[EMBED];             // user embedding (f16)
    __shared__ float sh_uvf[EMBED];
    __shared__ float sh_scores[EMBED];
    __shared__ float sh_att[EMBED];
    __shared__ float sh_agg[EMBED];

    const int b    = blockIdx.x;
    const int tid  = threadIdx.x;
    const int wave = tid >> 5;     // wave32
    const int lane = tid & 31;
    const int g    = lane >> 4;    // lane group (K half for A / B operands)
    const int ln   = lane & 15;
    const int len  = lengths[b];

    if (tid < EMBED) {
        float u = u2e[(size_t)nodes[b] * EMBED + tid];
        sh_uvf[tid] = u;
        sh_uvh[tid] = (_Float16)u;
        sh_agg[tid] = 0.0f;
    }
    __syncthreads();

    // Row handled by this lane (lanes L and L+16 cover the same row m=ln,
    // different K halves, per the 16-bit A-matrix layout). Padding rows
    // (l >= HL) just reuse row 0 indices: they compute finite garbage whose
    // attention weight is exactly 0 after the masked softmax.
    const int l    = wave * 16 + ln;
    const int item = (l < HL) ? hist_uv[b * HL + l] : 0;
    const int rate = (l < HL) ? hist_r [b * HL + l] : 0;

    const v8f vzero = {0.f, 0.f, 0.f, 0.f, 0.f, 0.f, 0.f, 0.f};

    // A operand from a 32-wide f32 K-slice (global): two 8-float runs.
    auto loadA_f32 = [&](const float* slice) -> v16h {
        float4 r0 = *reinterpret_cast<const float4*>(slice + g * 8);
        float4 r1 = *reinterpret_cast<const float4*>(slice + g * 8 + 4);
        float4 r2 = *reinterpret_cast<const float4*>(slice + 16 + g * 8);
        float4 r3 = *reinterpret_cast<const float4*>(slice + 16 + g * 8 + 4);
        V16H A;
        A.h[0]  = (_Float16)r0.x; A.h[1]  = (_Float16)r0.y;
        A.h[2]  = (_Float16)r0.z; A.h[3]  = (_Float16)r0.w;
        A.h[4]  = (_Float16)r1.x; A.h[5]  = (_Float16)r1.y;
        A.h[6]  = (_Float16)r1.z; A.h[7]  = (_Float16)r1.w;
        A.h[8]  = (_Float16)r2.x; A.h[9]  = (_Float16)r2.y;
        A.h[10] = (_Float16)r2.z; A.h[11] = (_Float16)r2.w;
        A.h[12] = (_Float16)r3.x; A.h[13] = (_Float16)r3.y;
        A.h[14] = (_Float16)r3.z; A.h[15] = (_Float16)r3.w;
        return A.v;
    };

    // A operand from a staged f16 row (LDS): two 16-byte chunks.
    const _Float16* stage_row = sh_stage + (wave * 16 + ln) * EMBED;
    const _Float16* O_row     = sh_O     + (wave * 16 + ln) * EMBED;
    auto loadA_lds = [&](const _Float16* rowbase, int kb) -> v16h {
        V16H A;
        A.q[0] = *reinterpret_cast<const uint4*>(rowbase + kb * 32 + g * 8);
        A.q[1] = *reinterpret_cast<const uint4*>(rowbase + kb * 32 + 16 + g * 8);
        return A.v;
    };

    // ---------------- layer 1: relu([e_uv|e_r] @ w1 + b1) ----------------
    {
        const float* vrow = v2e + (size_t)item * EMBED;
        const float* rrow = r2e + (size_t)rate * EMBED;
        V8F acc[4];
        #pragma unroll
        for (int nt = 0; nt < 4; ++nt) acc[nt].v = vzero;
        #pragma unroll
        for (int kb = 0; kb < 4; ++kb) {
            v16h A = loadA_f32((kb < 2) ? (vrow + kb * 32) : (rrow + (kb - 2) * 32));
            #pragma unroll
            for (int nt = 0; nt < 4; ++nt) {
                v16h B = *reinterpret_cast<const v16h*>(w1h + ((kb * 4 + nt) * 32 + lane) * 16);
                acc[nt].v = __builtin_amdgcn_wmma_f32_16x16x32_f16(
                    false, A, false, B, (short)0, acc[nt].v, false, false);
            }
        }
        #pragma unroll
        for (int nt = 0; nt < 4; ++nt) {
            const int n  = nt * 16 + ln;
            const float bb = b1[n];
            #pragma unroll
            for (int v = 0; v < 8; ++v) {
                float y = fmaxf(acc[nt].f[v] + bb, 0.f);
                sh_stage[(wave * 16 + 8 * g + v) * EMBED + n] = (_Float16)y;
            }
        }
    }
    __syncthreads();

    // ---------------- layer 2: o_history = relu(Y @ w2 + b2) ----------------
    V8F Of[4];  // o_history rows in f32 C-layout, kept live for aggregation
    {
        V8F acc[4];
        #pragma unroll
        for (int nt = 0; nt < 4; ++nt) acc[nt].v = vzero;
        #pragma unroll
        for (int kb = 0; kb < 2; ++kb) {
            v16h A = loadA_lds(stage_row, kb);
            #pragma unroll
            for (int nt = 0; nt < 4; ++nt) {
                v16h B = *reinterpret_cast<const v16h*>(w2h + ((kb * 4 + nt) * 32 + lane) * 16);
                acc[nt].v = __builtin_amdgcn_wmma_f32_16x16x32_f16(
                    false, A, false, B, (short)0, acc[nt].v, false, false);
            }
        }
        #pragma unroll
        for (int nt = 0; nt < 4; ++nt) {
            const int n  = nt * 16 + ln;
            const float bb = b2[n];
            #pragma unroll
            for (int v = 0; v < 8; ++v) {
                float o = fmaxf(acc[nt].f[v] + bb, 0.f);
                Of[nt].f[v] = o;
                sh_O[(wave * 16 + 8 * g + v) * EMBED + n] = (_Float16)o;
            }
        }
    }
    __syncthreads();

    // ------------- attention layer 1: relu([O | uv] @ a1 + ba1) -------------
    {
        V8F acc[4];
        #pragma unroll
        for (int nt = 0; nt < 4; ++nt) acc[nt].v = vzero;
        #pragma unroll
        for (int kb = 0; kb < 4; ++kb) {
            v16h A = (kb < 2) ? loadA_lds(O_row, kb) : loadA_lds(sh_uvh, kb - 2);
            #pragma unroll
            for (int nt = 0; nt < 4; ++nt) {
                v16h B = *reinterpret_cast<const v16h*>(a1h + ((kb * 4 + nt) * 32 + lane) * 16);
                acc[nt].v = __builtin_amdgcn_wmma_f32_16x16x32_f16(
                    false, A, false, B, (short)0, acc[nt].v, false, false);
            }
        }
        #pragma unroll
        for (int nt = 0; nt < 4; ++nt) {
            const int n  = nt * 16 + ln;
            const float bb = ba1[n];
            #pragma unroll
            for (int v = 0; v < 8; ++v) {
                float y = fmaxf(acc[nt].f[v] + bb, 0.f);
                sh_stage[(wave * 16 + 8 * g + v) * EMBED + n] = (_Float16)y;
            }
        }
    }
    __syncthreads();

    // ----- attention layer 2 + scores: (relu(h@a2+ba2)) @ a3 + ba3 -----
    {
        V8F acc[4];
        #pragma unroll
        for (int nt = 0; nt < 4; ++nt) acc[nt].v = vzero;
        #pragma unroll
        for (int kb = 0; kb < 2; ++kb) {
            v16h A = loadA_lds(stage_row, kb);
            #pragma unroll
            for (int nt = 0; nt < 4; ++nt) {
                v16h B = *reinterpret_cast<const v16h*>(a2h + ((kb * 4 + nt) * 32 + lane) * 16);
                acc[nt].v = __builtin_amdgcn_wmma_f32_16x16x32_f16(
                    false, A, false, B, (short)0, acc[nt].v, false, false);
            }
        }
        float part[8];
        #pragma unroll
        for (int v = 0; v < 8; ++v) part[v] = 0.f;
        #pragma unroll
        for (int nt = 0; nt < 4; ++nt) {
            const int n  = nt * 16 + ln;
            const float bb = ba2[n];
            const float av = a3[n];
            #pragma unroll
            for (int v = 0; v < 8; ++v)
                part[v] += fmaxf(acc[nt].f[v] + bb, 0.f) * av;
        }
        // Sum over the 16 lanes of each group (xor masks stay inside group).
        #pragma unroll
        for (int v = 0; v < 8; ++v) {
            #pragma unroll
            for (int off = 1; off < 16; off <<= 1)
                part[v] += __shfl_xor(part[v], off);
        }
        const float ba3v = ba3[0];
        if (ln == 0) {
            #pragma unroll
            for (int v = 0; v < 8; ++v)
                sh_scores[wave * 16 + 8 * g + v] = part[v] + ba3v;
        }
    }
    __syncthreads();

    // ---------------- masked softmax over history (wave 0) ----------------
    if (wave == 0) {
        const int l0 = lane, l1 = lane + 32;
        float s0 = (l0 < len) ? sh_scores[l0] : -3.0e38f;
        float s1 = (l1 < len) ? sh_scores[l1] : -3.0e38f;
        float m = fmaxf(s0, s1);
        #pragma unroll
        for (int off = 1; off < 32; off <<= 1) m = fmaxf(m, __shfl_xor(m, off));
        float e0 = (l0 < len) ? __expf(s0 - m) : 0.f;
        float e1 = (l1 < len) ? __expf(s1 - m) : 0.f;
        float s = e0 + e1;
        #pragma unroll
        for (int off = 1; off < 32; off <<= 1) s += __shfl_xor(s, off);
        float inv = 1.0f / s;
        sh_att[l0] = e0 * inv;
        sh_att[l1] = e1 * inv;
    }
    __syncthreads();

    // ---------------- agg = sum_l att[l] * o_history[l] ----------------
    #pragma unroll
    for (int nt = 0; nt < 4; ++nt) {
        float p = 0.f;
        #pragma unroll
        for (int v = 0; v < 8; ++v)
            p += sh_att[wave * 16 + 8 * g + v] * Of[nt].f[v];
        p += __shfl_xor(p, 16);  // combine the two row-halves (same column)
        if (g == 0) atomicAdd(&sh_agg[nt * 16 + ln], p);
    }
    __syncthreads();

    if (tid < EMBED)
        out[(size_t)b * EMBED + tid] = (len > 0) ? sh_agg[tid] : sh_uvf[tid];
}

// -----------------------------------------------------------------------------
extern "C" void kernel_launch(void* const* d_in, const int* in_sizes, int n_in,
                              void* d_out, int out_size, void* d_ws, size_t ws_size,
                              hipStream_t stream) {
    const int*   nodes   = (const int*)  d_in[0];
    const int*   hist_uv = (const int*)  d_in[1];
    const int*   hist_r  = (const int*)  d_in[2];
    const int*   lengths = (const int*)  d_in[3];
    const float* u2e     = (const float*)d_in[4];
    const float* v2e     = (const float*)d_in[5];
    const float* r2e     = (const float*)d_in[6];
    const float* w1      = (const float*)d_in[7];
    const float* b1      = (const float*)d_in[8];
    const float* w2      = (const float*)d_in[9];
    const float* b2      = (const float*)d_in[10];
    const float* a1      = (const float*)d_in[11];
    const float* ba1     = (const float*)d_in[12];
    const float* a2      = (const float*)d_in[13];
    const float* ba2     = (const float*)d_in[14];
    const float* a3      = (const float*)d_in[15];
    const float* ba3     = (const float*)d_in[16];

    _Float16* ws  = (_Float16*)d_ws;
    _Float16* w1h = ws;                 // 8192 halves (128x64)
    _Float16* w2h = ws + 8192;          // 4096 halves (64x64)
    _Float16* a1h = ws + 12288;         // 8192 halves (128x64)
    _Float16* a2h = ws + 20480;         // 4096 halves (64x64)

    // Re-swizzle weights every call (workspace is not preserved).
    cvt_weights<<<16, 256, 0, stream>>>(w1, w1h, 4096);
    cvt_weights<<<8,  256, 0, stream>>>(w2, w2h, 2048);
    cvt_weights<<<16, 256, 0, stream>>>(a1, a1h, 4096);
    cvt_weights<<<8,  256, 0, stream>>>(a2, a2h, 2048);

    const int B = in_sizes[0];  // 4096 batch elements
    uv_agg<<<B, 128, 0, stream>>>(nodes, hist_uv, hist_r, lengths,
                                  u2e, v2e, r2e,
                                  w1h, b1, w2h, b2, a1h, ba1, a2h, ba2,
                                  a3, ba3, (float*)d_out);
}
